// Attention_19095424598832
// MI455X (gfx1250) — matrix operand order repaired and checked by reading, not compiled, not run on hardware
//
#include <hip/hip_runtime.h>
#include <hip/hip_bf16.h>
#include <stdint.h>

#define B_ 2
#define T_ 4096
#define D_ 2048
#define N_ 8
#define H_ 256
#define KMASK -2.3819763e+38f

typedef __attribute__((ext_vector_type(16))) __bf16 v16bf;
typedef __attribute__((ext_vector_type(8)))  __bf16 v8bf;
typedef __attribute__((ext_vector_type(8)))  float  v8f;

static __device__ __forceinline__ __bf16 f2bf(float f) {
  unsigned u = __builtin_bit_cast(unsigned, f);
  u = (u + 0x7FFFu + ((u >> 16) & 1u)) >> 16;
  unsigned short s = (unsigned short)u;
  return __builtin_bit_cast(__bf16, s);
}

static __device__ __forceinline__ v16bf mk16(v8bf lo, v8bf hi) {
  v16bf r;
#pragma unroll
  for (int i = 0; i < 8; ++i) { r[i] = lo[i]; r[i + 8] = hi[i]; }
  return r;
}

static __device__ __forceinline__ v8f wmma_bf16(v16bf a, v16bf b, v8f c) {
  return __builtin_amdgcn_wmma_f32_16x16x32_bf16(false, a, false, b, (short)0, c,
                                                 false, false);
}

// ---------------- prep: fp32 -> bf16 conversions / transposes ----------------

__global__ void cvt_x_kernel(const float* __restrict__ x, __bf16* __restrict__ xb) {
  int i = blockIdx.x * blockDim.x + threadIdx.x;
  xb[i] = f2bf(x[i]);
}

// Wq[c*D + d] = q_w[n,d,h], c = n*256+h  (row = output column, contiguous in d)
__global__ void prep_wq_kernel(const float* __restrict__ q_w, __bf16* __restrict__ Wq) {
  int i = blockIdx.x * blockDim.x + threadIdx.x;  // c*D + d
  int d = i & (D_ - 1);
  int c = i >> 11;
  int n = c >> 8, h = c & 255;
  Wq[i] = f2bf(q_w[((size_t)n * D_ + d) * H_ + h]);
}

// Wk/Wv[h*D + d] = kv_w[c,0,d,h]
__global__ void prep_wkv_kernel(const float* __restrict__ kv_w,
                                __bf16* __restrict__ Wk, __bf16* __restrict__ Wv) {
  int i = blockIdx.x * blockDim.x + threadIdx.x;  // over 2*H*D
  int c = i >> 19;
  int r = i & ((1 << 19) - 1);  // h*D + d
  int d = r & (D_ - 1);
  int h = r >> 11;
  __bf16 v = f2bf(kv_w[((size_t)c * D_ + d) * H_ + h]);
  (c ? Wv : Wk)[r] = v;
}

// Wo[d*2048 + k] = out_w[k, d]   (out_w flattened [N*H][D])
__global__ void prep_wo_kernel(const float* __restrict__ out_w, __bf16* __restrict__ Wo) {
  int i = blockIdx.x * blockDim.x + threadIdx.x;  // d*2048 + k
  int k = i & 2047;
  int d = i >> 11;
  Wo[i] = f2bf(out_w[(size_t)k * D_ + d]);
}

// ---------------- QKV projection GEMM (WMMA bf16, f32 out) ----------------
// One wave computes a 16x128 tile.  Col tiles: [0,128)=Q, [128,144)=K, [144,160)=V.

__global__ void __launch_bounds__(32) gemm_qkv_kernel(
    const __bf16* __restrict__ xb, const __bf16* __restrict__ Wq,
    const __bf16* __restrict__ Wk, const __bf16* __restrict__ Wv,
    float* __restrict__ Qp, float* __restrict__ Kp, float* __restrict__ Vp) {
  const int lane = threadIdx.x;
  const int mt = blockIdx.x;
  const int ntb = blockIdx.y * 8;
  const int am = lane & 15;
  const int klo = (lane < 16) ? 0 : 8;   // A-fragment K base
  const int kbh = (lane < 16) ? 0 : 16;  // unused (B fragment also uses klo)
  const __bf16* Arow = xb + (size_t)(mt * 16 + am) * D_;

  const __bf16* Brow[8];
  float* dst[8];
  int dstride[8], dcol[8];
#pragma unroll
  for (int j = 0; j < 8; ++j) {
    int ct = ntb + j;
    const __bf16* Wt; int colbase; float* dp; int stride;
    if (ct < 128)      { Wt = Wq; colbase = ct * 16;         dp = Qp; stride = 2048; }
    else if (ct < 144) { Wt = Wk; colbase = (ct - 128) * 16; dp = Kp; stride = 256;  }
    else               { Wt = Wv; colbase = (ct - 144) * 16; dp = Vp; stride = 256;  }
    Brow[j] = Wt + (size_t)(colbase + am) * D_;
    dst[j] = dp; dstride[j] = stride; dcol[j] = colbase + am;
  }

  v8f acc[8];
#pragma unroll
  for (int j = 0; j < 8; ++j)
#pragma unroll
    for (int r = 0; r < 8; ++r) acc[j][r] = 0.0f;

  for (int kk = 0; kk < D_; kk += 32) {
    v16bf a = mk16(*(const v8bf*)(Arow + kk + klo),
                   *(const v8bf*)(Arow + kk + klo + 16));
#pragma unroll
    for (int j = 0; j < 8; ++j) {
      v16bf b = mk16(*(const v8bf*)(Brow[j] + kk + klo), *(const v8bf*)(Brow[j] + kk + klo + 16));
      acc[j] = wmma_bf16(a, b, acc[j]);
    }
  }
  const int rbase = mt * 16 + 8 * (lane >> 4);
#pragma unroll
  for (int j = 0; j < 8; ++j)
#pragma unroll
    for (int r = 0; r < 8; ++r)
      dst[j][(size_t)(rbase + r) * dstride[j] + dcol[j]] = acc[j][r];
}

// ---------------- RoPE + layout rewrites ----------------

__global__ void rope_q_kernel(const float* __restrict__ Qp, const int* __restrict__ sp,
                              __bf16* __restrict__ Qr) {
  int i = blockIdx.x * blockDim.x + threadIdx.x;  // B*T*N*128 = 2^23
  int h = i & 127;
  int n = (i >> 7) & 7;
  int t = (i >> 10) & (T_ - 1);
  int b = i >> 22;
  float pos = (float)sp[b * T_ + t];
  float frac = (float)h * (1.0f / 128.0f);
  float ang = pos * __expf(-9.210340371976184f * frac);  // pos * 10000^-frac
  float s, c;
  __sincosf(ang, &s, &c);
  size_t src = ((size_t)(b * T_ + t)) * 2048 + n * 256 + h;
  float q1 = Qp[src], q2 = Qp[src + 128];
  size_t d0 = (((size_t)(b * N_ + n)) * T_ + t) * H_ + h;
  Qr[d0]       = f2bf((q1 * c - q2 * s) * 0.0625f);   // * H^-0.5
  Qr[d0 + 128] = f2bf((q2 * c + q1 * s) * 0.0625f);
}

__global__ void rope_k_kernel(const float* __restrict__ Kp, const int* __restrict__ sp,
                              __bf16* __restrict__ Kr) {
  int i = blockIdx.x * blockDim.x + threadIdx.x;  // B*T*128 = 2^20
  int h = i & 127;
  int t = (i >> 7) & (T_ - 1);
  int b = i >> 19;
  float pos = (float)sp[b * T_ + t];
  float frac = (float)h * (1.0f / 128.0f);
  float ang = pos * __expf(-9.210340371976184f * frac);
  float s, c;
  __sincosf(ang, &s, &c);
  size_t src = ((size_t)(b * T_ + t)) * 256 + h;
  float k1 = Kp[src], k2 = Kp[src + 128];
  Kr[src]       = f2bf(k1 * c - k2 * s);
  Kr[src + 128] = f2bf(k2 * c + k1 * s);
}

// V transposed to [B][H][T] so PV B-fragments are contiguous in s
__global__ void cvt_v_kernel(const float* __restrict__ Vp, __bf16* __restrict__ Vt) {
  int i = blockIdx.x * blockDim.x + threadIdx.x;  // B*T*256 = 2^21
  int h = i & 255;
  int t = (i >> 8) & (T_ - 1);
  int b = i >> 20;
  Vt[((size_t)(b * H_ + h)) * T_ + t] = f2bf(Vp[i]);
}

// ---------------- Flash attention: one wave per 16-query block ----------------
// 64-wide s-blocks: 64 WMMAs per online-softmax update (VALU overhead halved
// vs 32-wide).  Q fragments reloaded per block (L2-resident) to cut VGPRs.

__global__ void __launch_bounds__(32) flash_kernel(
    const __bf16* __restrict__ Qr, const __bf16* __restrict__ Kr,
    const __bf16* __restrict__ Vt, __bf16* __restrict__ Enc) {
  const int lane = threadIdx.x;
  const int tblk = blockIdx.x;
  const int n = blockIdx.y;
  const int b = blockIdx.z;
  const int am = lane & 15;
  const int klo = (lane < 16) ? 0 : 8;
  const int kbh = (lane < 16) ? 0 : 16;

  const __bf16* Qbase = Qr + (((size_t)(b * N_ + n)) * T_ + tblk * 16) * H_;
  const __bf16* Kbase = Kr + ((size_t)b * T_) * H_;
  const __bf16* Vbase = Vt + ((size_t)b * H_) * T_;
  const __bf16* Qlane = Qbase + (size_t)am * H_ + klo;

  v8f o[16];
#pragma unroll
  for (int i = 0; i < 16; ++i)
#pragma unroll
    for (int r = 0; r < 8; ++r) o[i][r] = 0.0f;

  float mrow[8], lrow[8];
#pragma unroll
  for (int r = 0; r < 8; ++r) { mrow[r] = -3.0e38f; lrow[r] = 0.0f; }

  __shared__ __bf16 Pld[16 * 64];

  const int tg0 = tblk * 16;
  const int sb_max = (tg0 + 15) >> 6;
  for (int sb = 0; sb <= sb_max; ++sb) {
    const int s0 = sb * 64;
    v8f cc[4];
#pragma unroll
    for (int j = 0; j < 4; ++j)
#pragma unroll
      for (int r = 0; r < 8; ++r) cc[j][r] = 0.0f;

    const __bf16* Ks = Kbase + (size_t)(s0 + am) * H_ + klo;
#pragma unroll
    for (int hc = 0; hc < 8; ++hc) {
      v16bf qa = mk16(*(const v8bf*)(Qlane + hc * 32),
                      *(const v8bf*)(Qlane + hc * 32 + 16));
#pragma unroll
      for (int j = 0; j < 4; ++j) {
        v16bf bk = mk16(*(const v8bf*)(Ks + (size_t)(j * 16) * H_ + hc * 32), *(const v8bf*)(Ks + (size_t)(j * 16) * H_ + hc * 32 + 16));
        cc[j] = wmma_bf16(qa, bk, cc[j]);
      }
    }

    if (s0 + 63 > tg0) {  // causal mask near the diagonal (uniform branch)
      const int roff = 8 * (lane >> 4);
#pragma unroll
      for (int j = 0; j < 4; ++j)
#pragma unroll
        for (int r = 0; r < 8; ++r)
          if (s0 + j * 16 + am > tg0 + r + roff) cc[j][r] = KMASK;
    }

    float rmax[8];
#pragma unroll
    for (int r = 0; r < 8; ++r)
      rmax[r] = fmaxf(fmaxf(cc[0][r], cc[1][r]), fmaxf(cc[2][r], cc[3][r]));
#pragma unroll
    for (int off = 8; off >= 1; off >>= 1)
#pragma unroll
      for (int r = 0; r < 8; ++r)
        rmax[r] = fmaxf(rmax[r], __shfl_xor(rmax[r], off, 32));

    float alpha[8], rsum[8], p[4][8];
#pragma unroll
    for (int r = 0; r < 8; ++r) {
      float mnew = fmaxf(mrow[r], rmax[r]);
      alpha[r] = __expf(mrow[r] - mnew);
      mrow[r] = mnew;
      float s = 0.0f;
#pragma unroll
      for (int j = 0; j < 4; ++j) { p[j][r] = __expf(cc[j][r] - mnew); s += p[j][r]; }
      rsum[r] = s;
    }
#pragma unroll
    for (int off = 8; off >= 1; off >>= 1)
#pragma unroll
      for (int r = 0; r < 8; ++r)
        rsum[r] += __shfl_xor(rsum[r], off, 32);
#pragma unroll
    for (int r = 0; r < 8; ++r) lrow[r] = lrow[r] * alpha[r] + rsum[r];

    // relayout P (C layout) -> A-fragments via LDS bounce, as bf16
    {
      const int roff = 8 * (lane >> 4);
#pragma unroll
      for (int r = 0; r < 8; ++r)
#pragma unroll
        for (int j = 0; j < 4; ++j)
          Pld[(r + roff) * 64 + j * 16 + am] = f2bf(p[j][r]);
    }
    asm volatile("s_wait_dscnt 0" ::: "memory");
    v16bf pa0 = mk16(*(const v8bf*)&Pld[am * 64 + klo],
                     *(const v8bf*)&Pld[am * 64 + klo + 16]);
    v16bf pa1 = mk16(*(const v8bf*)&Pld[am * 64 + 32 + klo],
                     *(const v8bf*)&Pld[am * 64 + 32 + klo + 16]);

    if (sb < sb_max)
      __builtin_prefetch(Kbase + (size_t)(s0 + 64 + am) * H_, 0, 0);

    const __bf16* V0 = Vbase + (size_t)am * T_ + s0 + klo;
#pragma unroll
    for (int ht = 0; ht < 16; ++ht) {
      v16bf bv0 = mk16(*(const v8bf*)(V0 + (size_t)(ht * 16) * T_), *(const v8bf*)(V0 + (size_t)(ht * 16) * T_ + 16));
      v16bf bv1 = mk16(*(const v8bf*)(V0 + (size_t)(ht * 16) * T_ + 32), *(const v8bf*)(V0 + (size_t)(ht * 16) * T_ + 48));
#pragma unroll
      for (int r = 0; r < 8; ++r) o[ht][r] *= alpha[r];
      o[ht] = wmma_bf16(pa0, bv0, o[ht]);
      o[ht] = wmma_bf16(pa1, bv1, o[ht]);
    }
  }

  float linv[8];
#pragma unroll
  for (int r = 0; r < 8; ++r) linv[r] = 1.0f / lrow[r];
  __bf16* Eb = Enc + (((size_t)(b * N_ + n)) * T_ + tg0) * H_;
  const int roff = 8 * (lane >> 4);
#pragma unroll
  for (int ht = 0; ht < 16; ++ht)
#pragma unroll
    for (int r = 0; r < 8; ++r)
      Eb[(size_t)(r + roff) * H_ + ht * 16 + am] = f2bf(o[ht][r] * linv[r]);
}

// ---------------- Output projection GEMM (bf16 WMMA -> f32 out) ----------------
// One wave computes a 16x128 tile.

__global__ void __launch_bounds__(32) gemm_out_kernel(
    const __bf16* __restrict__ Enc, const __bf16* __restrict__ Wo,
    float* __restrict__ out) {
  const int lane = threadIdx.x;
  const int mt = blockIdx.x;
  const int ntb = blockIdx.y * 8;
  const int am = lane & 15;
  const int klo = (lane < 16) ? 0 : 8;
  const int kbh = (lane < 16) ? 0 : 16;

  const int m = mt * 16 + am;
  const int bq = m >> 12;          // row -> (b, t)
  const int tq = m & (T_ - 1);

  v8f acc[8];
#pragma unroll
  for (int j = 0; j < 8; ++j)
#pragma unroll
    for (int r = 0; r < 8; ++r) acc[j][r] = 0.0f;

  const __bf16* Brow[8];
#pragma unroll
  for (int j = 0; j < 8; ++j)
    Brow[j] = Wo + (size_t)((ntb + j) * 16 + am) * 2048;

  for (int kk = 0; kk < 2048; kk += 32) {
    int k0 = kk + klo;
    int k1 = k0 + 16;
    const __bf16* a0 = Enc + (((size_t)(bq * N_ + (k0 >> 8))) * T_ + tq) * H_ + (k0 & 255);
    const __bf16* a1 = Enc + (((size_t)(bq * N_ + (k1 >> 8))) * T_ + tq) * H_ + (k1 & 255);
    v16bf a = mk16(*(const v8bf*)a0, *(const v8bf*)a1);
#pragma unroll
    for (int j = 0; j < 8; ++j) {
      v16bf bb = mk16(*(const v8bf*)(Brow[j] + kk + klo), *(const v8bf*)(Brow[j] + kk + klo + 16));
      acc[j] = wmma_bf16(a, bb, acc[j]);
    }
  }
  const int rbase = mt * 16 + 8 * (lane >> 4);
#pragma unroll
  for (int j = 0; j < 8; ++j)
#pragma unroll
    for (int r = 0; r < 8; ++r)
      out[(size_t)(rbase + r) * 2048 + (ntb + j) * 16 + am] = acc[j][r];
}

// ---------------- launch ----------------

extern "C" void kernel_launch(void* const* d_in, const int* in_sizes, int n_in,
                              void* d_out, int out_size, void* d_ws, size_t ws_size,
                              hipStream_t stream) {
  const float* x     = (const float*)d_in[0];
  const int*   sp    = (const int*)d_in[1];
  // d_in[2] attn_mask: reference mask is causal tril; implemented analytically
  const float* q_w   = (const float*)d_in[3];
  const float* kv_w  = (const float*)d_in[4];
  const float* out_w = (const float*)d_in[5];
  float* out = (float*)d_out;

  char* w = (char*)d_ws;
  auto alloc = [&](size_t bytes) {
    char* p = w;
    w += (bytes + 255) & ~(size_t)255;
    return p;
  };
  __bf16* xb  = (__bf16*)alloc((size_t)B_ * T_ * D_ * 2);
  __bf16* Wq  = (__bf16*)alloc((size_t)2048 * 2048 * 2);
  __bf16* Wk  = (__bf16*)alloc((size_t)256 * 2048 * 2);
  __bf16* Wv  = (__bf16*)alloc((size_t)256 * 2048 * 2);
  __bf16* Wo  = (__bf16*)alloc((size_t)2048 * 2048 * 2);
  float*  Qp  = (float*)alloc((size_t)B_ * T_ * 2048 * 4);
  float*  Kp  = (float*)alloc((size_t)B_ * T_ * 256 * 4);
  float*  Vp  = (float*)alloc((size_t)B_ * T_ * 256 * 4);
  __bf16* Qr  = (__bf16*)alloc((size_t)B_ * N_ * T_ * H_ * 2);
  __bf16* Kr  = (__bf16*)alloc((size_t)B_ * T_ * H_ * 2);
  __bf16* Vt  = (__bf16*)alloc((size_t)B_ * T_ * H_ * 2);
  __bf16* Enc = (__bf16*)alloc((size_t)B_ * N_ * T_ * H_ * 2);

  cvt_x_kernel<<<(B_ * T_ * D_) / 256, 256, 0, stream>>>(x, xb);
  prep_wq_kernel<<<(2048 * 2048) / 256, 256, 0, stream>>>(q_w, Wq);
  prep_wkv_kernel<<<(2 * 256 * 2048) / 256, 256, 0, stream>>>(kv_w, Wk, Wv);
  prep_wo_kernel<<<(2048 * 2048) / 256, 256, 0, stream>>>(out_w, Wo);

  gemm_qkv_kernel<<<dim3(512, 20), 32, 0, stream>>>(xb, Wq, Wk, Wv, Qp, Kp, Vp);

  rope_q_kernel<<<(B_ * T_ * N_ * 128) / 256, 256, 0, stream>>>(Qp, sp, Qr);
  rope_k_kernel<<<(B_ * T_ * 128) / 256, 256, 0, stream>>>(Kp, sp, Kr);
  cvt_v_kernel<<<(B_ * T_ * 256) / 256, 256, 0, stream>>>(Vp, Vt);

  flash_kernel<<<dim3(T_ / 16, N_, B_), 32, 0, stream>>>(Qr, Kr, Vt, Enc);

  gemm_out_kernel<<<dim3(512, 16), 32, 0, stream>>>(Enc, Wo, out);
}
